// miniSiAll_8229157339224
// MI455X (gfx1250) — matrix operand order repaired and checked by reading, not compiled, not run on hardware
//
#include <hip/hip_runtime.h>

// ---------------------------------------------------------------------------
// Types / bf16 helpers (bit-exact, no reliance on __bf16 arithmetic support)
// ---------------------------------------------------------------------------
typedef __bf16 bf16_t;
typedef __bf16 v16bf __attribute__((ext_vector_type(16)));
typedef __bf16 v8bf  __attribute__((ext_vector_type(8)));
typedef float  v8f   __attribute__((ext_vector_type(8)));
typedef int    v4i   __attribute__((vector_size(16)));     // int4, matches builtin

#if defined(__has_builtin)
#if __has_builtin(__builtin_amdgcn_global_load_async_to_lds_b128)
#define HAVE_ASYNC_LDS 1
#endif
#endif

#ifdef HAVE_ASYNC_LDS
typedef __attribute__((address_space(1))) v4i glob_v4i;    // global int4
typedef __attribute__((address_space(3))) v4i lds_v4i;     // LDS int4
__device__ __forceinline__ void wait_async0() {
#if __has_builtin(__builtin_amdgcn_s_wait_asynccnt)
    __builtin_amdgcn_s_wait_asynccnt(0);
#else
    asm volatile("s_wait_asynccnt 0x0" ::: "memory");
#endif
}
#endif

__device__ __forceinline__ bf16_t f2b(float f) {
    unsigned u = __builtin_bit_cast(unsigned, f);
    unsigned r = (u + 0x7fffu + ((u >> 16) & 1u)) >> 16;   // RNE
    return __builtin_bit_cast(bf16_t, (unsigned short)r);
}
__device__ __forceinline__ bf16_t bzero() {
    return __builtin_bit_cast(bf16_t, (unsigned short)0);
}

// ---------------------------------------------------------------------------
// fp32 -> bf16 convert (lifted input)
// ---------------------------------------------------------------------------
__global__ void f32_to_bf16_kernel(const float* __restrict__ in,
                                   bf16_t* __restrict__ out, int n) {
    int i = blockIdx.x * 256 + threadIdx.x;
    if (i < n) out[i] = f2b(in[i]);
}

// ---------------------------------------------------------------------------
// Weight prep: [J][Cout][Cin][3][3] f32 -> [J][Cout][Kpad] bf16, zero-padded K
// ---------------------------------------------------------------------------
__global__ void prep_w_kernel(const float* __restrict__ w,
                              bf16_t* __restrict__ wb,
                              int JCout, int Ktot, int Kpad) {
    int i = blockIdx.x * 256 + threadIdx.x;
    if (i >= JCout * Kpad) return;
    int row = i / Kpad;
    int k   = i - row * Kpad;
    wb[i] = (k < Ktot) ? f2b(w[(long)row * Ktot + k]) : bzero();
}

// ---------------------------------------------------------------------------
// Scale-equivariant dilated 3x3 conv as implicit GEMM on v_wmma bf16.
// Workgroup tile: 32 Cout x 256 pixels (4 image rows), 8 waves, wave32.
// Each wave: 16 Cout rows x 4 pixel sub-tiles -> 4 WMMAs per K-chunk sharing
// one A fragment. Double-buffered LDS pipeline: chunk n+1's async weight copy
// (global_load_async_to_lds_b128) and coalesced im2col gather overlap chunk
// n's WMMAs; one s_barrier per chunk. LDS rows padded to 40 halves.
// ---------------------------------------------------------------------------
__global__ __launch_bounds__(256)
void scale_conv_wmma(const bf16_t* __restrict__ in, const bf16_t* __restrict__ wb,
                     bf16_t* __restrict__ out,
                     int Cout, int Ktot, int Kpad, int J,
                     long inBstride, long inSstride)
{
    constexpr int S   = 4;                     // NRATIO scale channels
    constexpr int LDA = 40;                    // padded halves per 32-k row
    __shared__ bf16_t lA[2][32 * LDA];         // 32 cout rows  x 32 k (+pad)
    __shared__ bf16_t lB[2][256 * LDA];        // 256 pixels    x 32 k (+pad)

    const int y4     = blockIdx.x * 4;         // first of 4 image rows
    const int coBase = blockIdx.y * 32;        // cout tile base
    const int bs     = blockIdx.z;             // b*S + s
    const int s      = bs & 3;
    const int b      = bs >> 2;
    const int d      = 1 << s;                 // dilation 2^s

    const int t    = threadIdx.x;
    const int lane = t & 31;
    const int wave = t >> 5;
    const int wm   = wave >> 2;                // 0..1 : cout 16-row group
    const int wn   = wave & 3;                 // 0..3 : pixel sub-tile base
    const int half = lane >> 4;
    const int l16  = lane & 15;

    const int Jeff   = (J < (S - s)) ? J : (S - s);
    const int nChunk = Kpad >> 5;
    const int total  = Jeff * nChunk;
    const long CoutKpad = (long)Cout * Kpad;

    // stage K-chunk (j, kc) into LDS slabs A/Bb
    auto stage = [&](bf16_t* A, bf16_t* Bb, int j, int kc) {
        const bf16_t* inP = in + inBstride * b + inSstride * (s + j);
        const bf16_t* wP  = wb + CoutKpad * j;
        // ---- A: 32 cout rows x 32 k of pre-padded weights ----
#ifdef HAVE_ASYNC_LDS
        if (wave < 4) {                        // 128 lanes x 16B = 2KB tile
            int r = t >> 2, seg = t & 3;       // row 0..31, 16B segment 0..3
            const bf16_t* src =
                wP + (long)(coBase + r) * Kpad + kc * 32 + seg * 8;
            __builtin_amdgcn_global_load_async_to_lds_b128(
                (glob_v4i*)(void*)src,
                (lds_v4i*)(void*)&A[r * LDA + seg * 8], 0, 0);
        }
#else
#pragma unroll
        for (int i = 0; i < 4; ++i) {
            int e = t + i * 256;               // < 1024
            int r = e >> 5, kk = e & 31;
            A[r * LDA + kk] = wP[(long)(coBase + r) * Kpad + kc * 32 + kk];
        }
#endif
        // ---- B: coalesced im2col, 256 px x 32 k; wave owns 4 k-indices ----
#pragma unroll
        for (int i = 0; i < 4; ++i) {
            int kk = wave * 4 + i;
            int k  = kc * 32 + kk;
            if (k < Ktot) {
                int ci  = k / 9;
                int rem = k - ci * 9;
                int ky  = rem / 3;
                int kx  = rem - ky * 3;
                const bf16_t* plane = inP + ((long)ci << 12);
                int iyOff = (ky - 1) * d;
                int ixOff = (kx - 1) * d;
#pragma unroll
                for (int pb = 0; pb < 8; ++pb) {
                    int p  = pb * 32 + lane;           // pixel 0..255
                    int iy = y4 + (pb >> 1) + iyOff;   // pixel row + dy
                    int ix = ((pb & 1) * 32 + lane) + ixOff;
                    bf16_t v = bzero();
                    if ((unsigned)iy < 64u && (unsigned)ix < 64u)
                        v = plane[(iy << 6) + ix];
                    Bb[p * LDA + kk] = v;
                }
            } else {
#pragma unroll
                for (int pb = 0; pb < 8; ++pb)
                    Bb[(pb * 32 + lane) * LDA + kk] = bzero();
            }
        }
    };

    v8f acc[4];
#pragma unroll
    for (int q = 0; q < 4; ++q)
#pragma unroll
        for (int i = 0; i < 8; ++i) acc[q][i] = 0.0f;

    // ---- pipelined main loop: stage(next) || wmma(current) ----
    int jn = 0, kcn = 0;                       // next chunk to stage
    stage(lA[0], lB[0], 0, 0);
    if (++kcn == nChunk) { kcn = 0; ++jn; }
#ifdef HAVE_ASYNC_LDS
    if (wave < 4) wait_async0();
#endif
    __syncthreads();

    for (int cur = 0; cur < total; ++cur) {
        const int  cb   = cur & 1;
        const bool more = (cur + 1 < total);
        if (more) {
            stage(lA[cb ^ 1], lB[cb ^ 1], jn, kcn);
            if (++kcn == nChunk) { kcn = 0; ++jn; }
        }

        // A fragment (shared by all 4 WMMAs of this wave)
        const int row = wm * 16 + l16;
        v8bf a0 = *(const v8bf*)&lA[cb][row * LDA + half * 8];
        v8bf a1 = *(const v8bf*)&lA[cb][row * LDA + 16 + half * 8];
        v16bf afrag;
#pragma unroll
        for (int i = 0; i < 8; ++i) { afrag[i] = a0[i]; afrag[8 + i] = a1[i]; }

#pragma unroll
        for (int q = 0; q < 4; ++q) {
            const int pix = (wn + q * 4) * 16 + l16;
            v8bf b0 = *(const v8bf*)&lB[cb][pix * LDA + half * 8];
            v8bf b1 = *(const v8bf*)&lB[cb][pix * LDA + 16 + half * 8];
            v16bf bfrag;
#pragma unroll
            for (int i = 0; i < 8; ++i) { bfrag[i] = b0[i]; bfrag[8 + i] = b1[i]; }
            acc[q] = __builtin_amdgcn_wmma_f32_16x16x32_bf16(
                false, afrag, false, bfrag, (short)0, acc[q], false, false);
        }

#ifdef HAVE_ASYNC_LDS
        if (more && wave < 4) wait_async0();   // next A tile landed
#endif
        __syncthreads();                       // publish next buffers / retire reads
    }

    // ---- epilogue: relu -> bf16; C/D layout: col=l16, row=half*8+r ----
#pragma unroll
    for (int q = 0; q < 4; ++q) {
        const int p = (wn + q * 4) * 16 + l16;
        const int y = y4 + (p >> 6);
        const int x = p & 63;
#pragma unroll
        for (int r = 0; r < 8; ++r) {
            int   co = coBase + wm * 16 + half * 8 + r;
            float v  = acc[q][r];
            v = v > 0.0f ? v : 0.0f;
            out[(((long)bs * Cout + co) << 12) + (y << 6) + x] = f2b(v);
        }
    }
}

// ---------------------------------------------------------------------------
// Per-(b,s,c) spatial mean over 64x64 (uint4-vectorized bf16 loads)
// ---------------------------------------------------------------------------
__global__ __launch_bounds__(256)
void spatial_mean_kernel(const bf16_t* __restrict__ h, float* __restrict__ meanC) {
    __shared__ float red[256];
    const uint4* p4 = (const uint4*)(h + ((long)blockIdx.x << 12));
    float sum = 0.0f;
#pragma unroll
    for (int it = 0; it < 2; ++it) {
        uint4 v = p4[threadIdx.x + 256 * it];
        unsigned w[4] = {v.x, v.y, v.z, v.w};
#pragma unroll
        for (int q = 0; q < 4; ++q) {
            sum += __builtin_bit_cast(float, w[q] << 16);
            sum += __builtin_bit_cast(float, w[q] & 0xffff0000u);
        }
    }
    red[threadIdx.x] = sum;
    __syncthreads();
    for (int o = 128; o > 0; o >>= 1) {
        if (threadIdx.x < o) red[threadIdx.x] += red[threadIdx.x + o];
        __syncthreads();
    }
    if (threadIdx.x == 0) meanC[blockIdx.x] = red[0] * (1.0f / 4096.0f);
}

// ---------------------------------------------------------------------------
// Head: project channel means with w5, pool scales with 4^{-s}
// ---------------------------------------------------------------------------
__global__ void head_kernel(const float* __restrict__ meanC,
                            const float* __restrict__ w5,
                            float* __restrict__ out, int B, int S, int C) {
    int idx = blockIdx.x * blockDim.x + threadIdx.x;
    if (idx >= B * 10) return;
    int b = idx / 10, f = idx - b * 10;
    float sum = 0.0f;
    for (int s = 0; s < S; ++s) {
        float fac = 1.0f / (float)(1 << (2 * s));     // RATIO^{-2s}
        float dot = 0.0f;
        const float* m = meanC + ((long)b * S + s) * C;
        for (int c = 0; c < C; ++c) dot += w5[f * C + c] * m[c];
        sum += fac * dot;
    }
    out[idx] = sum;
}

// ---------------------------------------------------------------------------
// Launch
// ---------------------------------------------------------------------------
extern "C" void kernel_launch(void* const* d_in, const int* in_sizes, int n_in,
                              void* d_out, int out_size, void* d_ws, size_t ws_size,
                              hipStream_t stream) {
    const float* x  = (const float*)d_in[0];
    const float* w1 = (const float*)d_in[1];
    const float* w2 = (const float*)d_in[2];
    const float* w3 = (const float*)d_in[3];
    const float* w4 = (const float*)d_in[4];
    const float* w5 = (const float*)d_in[5];
    float* out = (float*)d_out;
    (void)in_sizes; (void)n_in; (void)out_size; (void)ws_size;

    const int B = 32, S = 4, HW = 4096;
    char* ws = (char*)d_ws;
    size_t off = 0;
    auto alloc = [&](size_t bytes) -> char* {
        char* p = ws + off;
        off += (bytes + 255) & ~(size_t)255;
        return p;
    };
    bf16_t* xb    = (bf16_t*)alloc((size_t)B * 3 * HW * 2);
    bf16_t* wb1   = (bf16_t*)alloc((size_t)1 * 96 * 32 * 2);
    bf16_t* wb2   = (bf16_t*)alloc((size_t)3 * 96 * 864 * 2);
    bf16_t* wb3   = (bf16_t*)alloc((size_t)3 * 192 * 864 * 2);
    bf16_t* wb4   = (bf16_t*)alloc((size_t)3 * 192 * 1728 * 2);
    float*  meanC = (float*)alloc((size_t)B * S * 192 * 4);
    bf16_t* buf0  = (bf16_t*)alloc((size_t)B * S * 192 * HW * 2);
    bf16_t* buf1  = (bf16_t*)alloc((size_t)B * S * 192 * HW * 2);

    { int n = B * 3 * HW;
      f32_to_bf16_kernel<<<(n + 255) / 256, 256, 0, stream>>>(x, xb, n); }
    prep_w_kernel<<<(1 * 96  * 32   + 255) / 256, 256, 0, stream>>>(w1, wb1, 1 * 96,  27,   32);
    prep_w_kernel<<<(3 * 96  * 864  + 255) / 256, 256, 0, stream>>>(w2, wb2, 3 * 96,  864,  864);
    prep_w_kernel<<<(3 * 192 * 864  + 255) / 256, 256, 0, stream>>>(w3, wb3, 3 * 192, 864,  864);
    prep_w_kernel<<<(3 * 192 * 1728 + 255) / 256, 256, 0, stream>>>(w4, wb4, 3 * 192, 1728, 1728);

    dim3 blk(256);
    // L1: lifted x (scale stride 0), Cin=3 -> Cout=96, K=27 padded to 32, J=1
    scale_conv_wmma<<<dim3(16, 96 / 32, B * S), blk, 0, stream>>>(
        xb, wb1, buf0, 96, 27, 32, 1, (long)3 * HW, 0L);
    // L2: 96 -> 96, J=3
    scale_conv_wmma<<<dim3(16, 96 / 32, B * S), blk, 0, stream>>>(
        buf0, wb2, buf1, 96, 864, 864, 3, (long)S * 96 * HW, (long)96 * HW);
    // L3: 96 -> 192, J=3
    scale_conv_wmma<<<dim3(16, 192 / 32, B * S), blk, 0, stream>>>(
        buf1, wb3, buf0, 192, 864, 864, 3, (long)S * 96 * HW, (long)96 * HW);
    // L4: 192 -> 192, J=3
    scale_conv_wmma<<<dim3(16, 192 / 32, B * S), blk, 0, stream>>>(
        buf0, wb4, buf1, 192, 1728, 1728, 3, (long)S * 192 * HW, (long)192 * HW);

    spatial_mean_kernel<<<B * S * 192, 256, 0, stream>>>(buf1, meanC);
    head_kernel<<<(B * 10 + 255) / 256, 256, 0, stream>>>(meanC, w5, out, B, S, 192);
}
